// BiLSTMCRF_81570018885946
// MI455X (gfx1250) — matrix operand order repaired and checked by reading, not compiled, not run on hardware
//
#include <hip/hip_runtime.h>
#include <math.h>

// Problem constants (from reference)
#define TT 256
#define BB 64
#define EE 256
#define HH 512
#define G4 2048   // 4*H
#define LL 9

typedef __attribute__((ext_vector_type(16))) __bf16 v16bf;
typedef __attribute__((ext_vector_type(8)))  float  v8f;

typedef unsigned short u16;

// ---------- helpers ----------

__device__ __forceinline__ u16 f2bf(float f) {
    unsigned int u = __float_as_uint(f);
    unsigned int r = u + 0x7FFFu + ((u >> 16) & 1u);   // round-to-nearest-even
    if ((u & 0x7F800000u) == 0x7F800000u) r = u;       // Inf/NaN: truncate
    return (u16)(r >> 16);
}

__device__ __forceinline__ float sigmoidf(float x) {
    return 1.0f / (1.0f + __expf(-x));
}

// A operand (16x32 bf16, M x K): lane m = lane&15; elems 0..7 = K[k0..k0+7],
// elems 8..15 = K[k0+16..k0+23], k0 = kbase + 8*hi   (ISA 7.12.2)
__device__ __forceinline__ v16bf load_a(const u16* row, int kbase, int hi) {
    int k0 = kbase + hi * 8;
    union { v16bf v; uint4 u[2]; } r;
    r.u[0] = *(const uint4*)(row + k0);
    r.u[1] = *(const uint4*)(row + k0 + 16);
    return r.v;
}

// B operand (32x16 bf16, K x N) where memory holds W[n][k] (row-major, we do x*W^T):
// lane n = lane&15; elems 0..7 = K[k0..k0+7], elems 8..15 = K[k0+16..k0+23], k0 = kbase + 8*hi
__device__ __forceinline__ v16bf load_b(const u16* row, int kbase, int hi) {
    int k0 = kbase + hi * 8;
    union { v16bf v; uint4 u[2]; } r;
    const uint4* p = (const uint4*)(row + k0);
    r.u[0] = p[0];
    r.u[1] = p[2];
    return r.v;
}

__device__ __forceinline__ v8f wmma_bf16(v16bf a, v16bf b, v8f c) {
    return __builtin_amdgcn_wmma_f32_16x16x32_bf16(false, a, false, b,
                                                   (short)0, c, false, false);
}

// ---------- setup kernels ----------

// embedding gather -> bf16 x[t*B+b][e]
__global__ void k_gather(const float* __restrict__ emb,
                         const int* __restrict__ ids,   // [B][T]
                         u16* __restrict__ x_bf) {
    int idx = blockIdx.x * blockDim.x + threadIdx.x;
    if (idx >= TT * BB * EE) return;
    int e  = idx % EE;
    int tb = idx / EE;
    int t  = tb / BB;
    int b  = tb % BB;
    int id = ids[b * TT + t];
    x_bf[idx] = f2bf(emb[(size_t)id * EE + e]);
}

__global__ void k_cvt(const float* __restrict__ src, u16* __restrict__ dst, int n) {
    int i = blockIdx.x * blockDim.x + threadIdx.x;
    if (i < n) dst[i] = f2bf(src[i]);
}

// w_out [9][1024] -> padded bf16 [16][1024] (rows 9..15 zero)
__global__ void k_cvt_wout(const float* __restrict__ w_out, u16* __restrict__ dst) {
    int i = blockIdx.x * blockDim.x + threadIdx.x;
    if (i >= 16 * 2 * HH) return;
    int l = i / (2 * HH);
    int k = i % (2 * HH);
    dst[i] = (l < LL) ? f2bf(w_out[l * 2 * HH + k]) : (u16)0;
}

__global__ void k_zero_hc(u16* __restrict__ h, float* __restrict__ c) {
    int i = blockIdx.x * blockDim.x + threadIdx.x;
    if (i >= 2 * BB * HH) return;
    h[i] = 0;
    c[i] = 0.0f;
}

// ---------- recurrent step: gate preactivations via WMMA ----------
// gates[dir][b][g] = sum_e x[time][b][e]*w_ih[g][e] + sum_h hprev[b][h]*w_hh[g][h]
// grid: (128 ntiles, 4 mtiles, 2 dirs), block: 32 (one wave per 16x16 tile)
__global__ void k_gates(const u16* __restrict__ x_bf,     // [T*B][E]
                        const u16* __restrict__ wih_bf,   // [2][G4][E]
                        const u16* __restrict__ whh_bf,   // [2][G4][H]
                        const u16* __restrict__ h_buf,    // [2][B][H]
                        float* __restrict__ gates,        // [2][B][G4]
                        int t) {
    const int lane  = threadIdx.x;
    const int nn    = lane & 15;
    const int hi    = lane >> 4;
    const int ntile = blockIdx.x;
    const int mtile = blockIdx.y;
    const int dir   = blockIdx.z;

    const int b = mtile * 16 + nn;        // A-row for this lane
    const int g = ntile * 16 + nn;        // B-col (gate idx) for this lane
    const int time = dir ? (TT - 1 - t) : t;

    const u16* Ax = x_bf  + (size_t)(time * BB + b) * EE;
    const u16* Bi = wih_bf + ((size_t)dir * G4 + g) * EE;
    const u16* Ah = h_buf + ((size_t)dir * BB + b) * HH;
    const u16* Bh = whh_bf + ((size_t)dir * G4 + g) * HH;

    v8f acc = {};
    #pragma unroll
    for (int k = 0; k < EE; k += 32)
        acc = wmma_bf16(load_a(Ax, k, hi), load_b(Bi, k, hi), acc);
    #pragma unroll
    for (int k = 0; k < HH; k += 32)
        acc = wmma_bf16(load_a(Ah, k, hi), load_b(Bh, k, hi), acc);

    float* out = gates + (size_t)dir * BB * G4;
    const int rbase = mtile * 16 + hi * 8;   // D row = 8*hi + r  (ISA 7.12.2)
    #pragma unroll
    for (int r = 0; r < 8; r++)
        out[(size_t)(rbase + r) * G4 + (ntile * 16 + nn)] = acc[r];
}

// ---------- LSTM cell elementwise update ----------
__global__ void k_update(const float* __restrict__ gates,  // [2][B][G4]
                         const float* __restrict__ b_ih_f, const float* __restrict__ b_hh_f,
                         const float* __restrict__ b_ih_b, const float* __restrict__ b_hh_b,
                         u16* __restrict__ h_buf,          // [2][B][H]
                         float* __restrict__ c_buf,        // [2][B][H]
                         u16* __restrict__ h_all,          // [2][T][B][H]
                         int t) {
    int idx = blockIdx.x * blockDim.x + threadIdx.x;
    if (idx >= 2 * BB * HH) return;
    int dir = idx / (BB * HH);
    int rem = idx % (BB * HH);
    int b   = rem / HH;
    int j   = rem % HH;

    const float* bi = dir ? b_ih_b : b_ih_f;
    const float* bh = dir ? b_hh_b : b_hh_f;
    const float* gg = gates + ((size_t)dir * BB + b) * G4;

    float i_ = gg[j]           + bi[j]           + bh[j];            // i
    float f_ = gg[HH + j]      + bi[HH + j]      + bh[HH + j];       // f
    float g_ = gg[2 * HH + j]  + bi[2 * HH + j]  + bh[2 * HH + j];   // g
    float o_ = gg[3 * HH + j]  + bi[3 * HH + j]  + bh[3 * HH + j];   // o

    float c = c_buf[idx];
    c = sigmoidf(f_) * c + sigmoidf(i_) * tanhf(g_);
    float h = sigmoidf(o_) * tanhf(c);
    c_buf[idx] = c;

    u16 hb = f2bf(h);
    h_buf[idx] = hb;
    int time = dir ? (TT - 1 - t) : t;                 // un-reverse backward dir
    h_all[(((size_t)dir * TT + time) * BB + b) * HH + j] = hb;
}

// ---------- emissions: [T*B,1024] @ w_out^T (padded to N=16) ----------
__global__ void k_emis(const u16* __restrict__ h_all,   // [2][T][B][H]
                       const u16* __restrict__ wout_bf, // [16][1024]
                       const float* __restrict__ b_out, // [9]
                       float* __restrict__ emissions) { // [B][T][L]
    const int lane = threadIdx.x;
    const int nn   = lane & 15;
    const int hi   = lane >> 4;
    const int mt   = blockIdx.x;            // 0..1023 over (T*B)/16

    const int mrow = mt * 16 + nn;          // flat tb = t*B + b
    const int tA = mrow / BB, bA = mrow % BB;
    const u16* Af = h_all + (((size_t)0 * TT + tA) * BB + bA) * HH;
    const u16* Ab = h_all + (((size_t)1 * TT + tA) * BB + bA) * HH;
    const u16* Bw = wout_bf + (size_t)nn * (2 * HH);

    v8f acc = {};
    #pragma unroll
    for (int k = 0; k < HH; k += 32)
        acc = wmma_bf16(load_a(Af, k, hi), load_b(Bw, k, hi), acc);
    #pragma unroll
    for (int k = 0; k < HH; k += 32)
        acc = wmma_bf16(load_a(Ab, k, hi), load_b(Bw + HH, k, hi), acc);

    if (nn < LL) {
        float bo = b_out[nn];
        #pragma unroll
        for (int r = 0; r < 8; r++) {
            int row = mt * 16 + hi * 8 + r;
            int t = row / BB, b = row % BB;
            emissions[((size_t)b * TT + t) * LL + nn] = acc[r] + bo;
        }
    }
}

// ---------- CRF log-likelihood: one wave per batch element ----------
__global__ void k_crf(const float* __restrict__ emissions, // [B][T][L]
                      const int* __restrict__ labels,      // [B][T]
                      const int* __restrict__ amask,       // [B][T]
                      const float* __restrict__ start,
                      const float* __restrict__ endt,
                      const float* __restrict__ trans,     // [L][L]
                      float* __restrict__ llh) {
    const int b    = blockIdx.x;
    const int lane = threadIdx.x;
    const bool act = lane < LL;
    const float NEG = -1.0e30f;

    float trc[LL];
    #pragma unroll
    for (int i = 0; i < LL; i++)
        trc[i] = act ? trans[i * LL + lane] : 0.0f;

    const float* em0 = emissions + (size_t)b * TT * LL;
    int lab0 = labels[b * TT];
    int tg0  = (lab0 == -100) ? 0 : lab0;
    float alpha = act ? (start[lane] + em0[lane]) : NEG;
    float score = start[tg0] + em0[tg0];          // uniform across lanes
    int prev = tg0;

    for (int t = 1; t < TT; t++) {
        const float* em = emissions + ((size_t)b * TT + t) * LL;
        int lab = labels[b * TT + t];
        int mk  = (lab != -100) && (amask[b * TT + t] > 0);
        int tg  = (lab == -100) ? 0 : lab;
        if (mk) { score += trans[prev * LL + tg] + em[tg]; prev = tg; }

        float vals[LL];
        float m = NEG;
        #pragma unroll
        for (int i = 0; i < LL; i++) {
            float ai = __shfl(alpha, i, 32);
            float v = ai + trc[i];
            vals[i] = v;
            m = fmaxf(m, v);
        }
        float s = 0.0f;
        #pragma unroll
        for (int i = 0; i < LL; i++) s += __expf(vals[i] - m);
        float nxt = m + __logf(s) + (act ? em[lane] : 0.0f);
        if (mk && act) alpha = nxt;
    }

    float numer = score + endt[prev];

    float v = act ? (alpha + endt[lane]) : NEG;
    float m = v;
    for (int off = 16; off; off >>= 1) m = fmaxf(m, __shfl_xor(m, off, 32));
    float e = (v > -1.0e29f) ? __expf(v - m) : 0.0f;
    for (int off = 16; off; off >>= 1) e += __shfl_xor(e, off, 32);
    float logz = m + __logf(e);

    if (lane == 0) llh[b] = numer - logz;
}

__global__ void k_final(const float* __restrict__ llh, float* __restrict__ out) {
    int lane = threadIdx.x;                   // 32 threads
    float s = llh[lane] + llh[lane + 32];
    for (int off = 16; off; off >>= 1) s += __shfl_xor(s, off, 32);
    if (lane == 0) out[0] = -s / (float)BB;
}

// ---------- host ----------

extern "C" void kernel_launch(void* const* d_in, const int* in_sizes, int n_in,
                              void* d_out, int out_size, void* d_ws, size_t ws_size,
                              hipStream_t stream) {
    const int*   ids     = (const int*)  d_in[0];
    const int*   amask   = (const int*)  d_in[1];
    const int*   labels  = (const int*)  d_in[2];
    const float* emb     = (const float*)d_in[3];
    const float* w_ih_f  = (const float*)d_in[4];
    const float* w_hh_f  = (const float*)d_in[5];
    const float* b_ih_f  = (const float*)d_in[6];
    const float* b_hh_f  = (const float*)d_in[7];
    const float* w_ih_b  = (const float*)d_in[8];
    const float* w_hh_b  = (const float*)d_in[9];
    const float* b_ih_b  = (const float*)d_in[10];
    const float* b_hh_b  = (const float*)d_in[11];
    const float* w_out   = (const float*)d_in[12];
    const float* b_out   = (const float*)d_in[13];
    const float* start_t = (const float*)d_in[14];
    const float* end_t   = (const float*)d_in[15];
    const float* trans   = (const float*)d_in[16];
    float* out = (float*)d_out;

    // workspace carve-up (256B aligned)
    size_t off = 0;
    auto carve = [&](size_t bytes) -> char* {
        char* p = (char*)d_ws + off;
        off = (off + bytes + 255) & ~(size_t)255;
        return p;
    };
    u16*   x_bf    = (u16*)  carve((size_t)TT * BB * EE * 2);        // 8 MB
    u16*   wih_bf  = (u16*)  carve((size_t)2 * G4 * EE * 2);         // 2 MB
    u16*   whh_bf  = (u16*)  carve((size_t)2 * G4 * HH * 2);         // 4 MB
    u16*   wout_bf = (u16*)  carve((size_t)16 * 2 * HH * 2);         // 32 KB
    u16*   h_buf   = (u16*)  carve((size_t)2 * BB * HH * 2);
    float* c_buf   = (float*)carve((size_t)2 * BB * HH * 4);
    float* gates   = (float*)carve((size_t)2 * BB * G4 * 4);         // 1 MB
    u16*   h_all   = (u16*)  carve((size_t)2 * TT * BB * HH * 2);    // 32 MB
    float* emis    = (float*)carve((size_t)BB * TT * LL * 4);
    float* llh     = (float*)carve((size_t)BB * 4);
    (void)ws_size;

    // setup
    {
        int n = TT * BB * EE;
        k_gather<<<(n + 255) / 256, 256, 0, stream>>>(emb, ids, x_bf);
    }
    k_cvt<<<(G4 * EE + 255) / 256, 256, 0, stream>>>(w_ih_f, wih_bf,            G4 * EE);
    k_cvt<<<(G4 * EE + 255) / 256, 256, 0, stream>>>(w_ih_b, wih_bf + G4 * EE,  G4 * EE);
    k_cvt<<<(G4 * HH + 255) / 256, 256, 0, stream>>>(w_hh_f, whh_bf,            G4 * HH);
    k_cvt<<<(G4 * HH + 255) / 256, 256, 0, stream>>>(w_hh_b, whh_bf + G4 * HH,  G4 * HH);
    k_cvt_wout<<<(16 * 2 * HH + 255) / 256, 256, 0, stream>>>(w_out, wout_bf);
    k_zero_hc<<<(2 * BB * HH + 255) / 256, 256, 0, stream>>>(h_buf, c_buf);

    // bidirectional LSTM recurrence: WMMA gates + elementwise cell per step
    dim3 ggrid(G4 / 16, BB / 16, 2);   // 128 x 4 x 2 one-wave tiles
    for (int t = 0; t < TT; t++) {
        k_gates<<<ggrid, 32, 0, stream>>>(x_bf, wih_bf, whh_bf, h_buf, gates, t);
        k_update<<<(2 * BB * HH + 255) / 256, 256, 0, stream>>>(
            gates, b_ih_f, b_hh_f, b_ih_b, b_hh_b, h_buf, c_buf, h_all, t);
    }

    // emissions projection (WMMA, N padded 9->16)
    k_emis<<<(TT * BB) / 16, 32, 0, stream>>>(h_all, wout_bf, b_out, emis);

    // CRF scan + final mean
    k_crf<<<BB, 32, 0, stream>>>(emis, labels, amask, start_t, end_t, trans, llh);
    k_final<<<1, 32, 0, stream>>>(llh, out);
}
